// AKLKA_48730698940697
// MI455X (gfx1250) — compile-verified, matrix operand order checked by reading, not run on hardware
//
#include <hip/hip_runtime.h>
#include <hip/hip_bf16.h>
#include <cstdint>

#define BB   8
#define CC   256
#define HHW  4096      // 64*64
#define PP   32768     // B*H*W
#define KAK  768       // C*3

typedef __attribute__((ext_vector_type(16))) _Float16 v16h;
typedef __attribute__((ext_vector_type(8)))  _Float16 v8h;
typedef __attribute__((ext_vector_type(8)))  float    v8f;

static __device__ __forceinline__ float clamp63(float v) {
  return fminf(fmaxf(v, 0.f), 63.f);
}

// ---------------------------------------------------------------- repack
__global__ __launch_bounds__(256) void repack_kernel(
    const float* __restrict__ ak_w, const float* __restrict__ lka1_w,
    const float* __restrict__ conv_w,
    _Float16* __restrict__ Aak, _Float16* __restrict__ Alk, _Float16* __restrict__ Acv)
{
  int tid = blockIdx.x * 256 + threadIdx.x;
  if (tid < 256 * 768) {
    int co = tid / 768, k = tid - co * 768;
    int n = k >> 8, ci = k & 255;                 // k = n*256 + ci
    Aak[co * 768 + k] = (_Float16)ak_w[(co * 256 + ci) * 3 + n];
  }
  if (tid < 65536) {
    Alk[tid] = (_Float16)lka1_w[tid];
    Acv[tid] = (_Float16)conv_w[tid];
  }
}

// ----------------------------------------------- 3x3 conv, 256 -> 6 (offsets)
__global__ __launch_bounds__(256) void pconv_kernel(
    const float* __restrict__ x, const float* __restrict__ p_w,
    const float* __restrict__ p_b, float* __restrict__ off)
{
  int p = blockIdx.x * 256 + threadIdx.x;         // pixel id
  int b = p >> 12, hw = p & 4095, h = hw >> 6, w0 = hw & 63;
  float acc[6];
#pragma unroll
  for (int o = 0; o < 6; ++o) acc[o] = p_b[o];
  for (int c = 0; c < CC; ++c) {
    const float* xp = x + (((size_t)b * CC + c) << 12);
    float v[9];
#pragma unroll
    for (int ky = 0; ky < 3; ++ky) {
      int yy = h + ky - 1;
#pragma unroll
      for (int kx = 0; kx < 3; ++kx) {
        int xx = w0 + kx - 1;
        v[ky * 3 + kx] = (yy >= 0 && yy < 64 && xx >= 0 && xx < 64) ? xp[yy * 64 + xx] : 0.f;
      }
    }
    const float* wp = p_w + c * 9;                // p_w[o][c][ky][kx] = o*2304 + c*9 + t
#pragma unroll
    for (int o = 0; o < 6; ++o) {
      const float* wo = wp + o * 2304;
      float a = acc[o];
#pragma unroll
      for (int t = 0; t < 9; ++t) a = fmaf(v[t], wo[t], a);
      acc[o] = a;
    }
  }
#pragma unroll
  for (int o = 0; o < 6; ++o) off[((size_t)(b * 6 + o) << 12) + hw] = acc[o];
}

// ------------------------------------------- bilinear indices + weights (p,n)
__global__ __launch_bounds__(256) void sprep_kernel(
    const float* __restrict__ off, int* __restrict__ idx, float* __restrict__ wgt)
{
  int id = blockIdx.x * 256 + threadIdx.x;        // 0 .. 3*PP-1
  int p = id / 3, n = id - p * 3;
  int b = p >> 12, hw = p & 4095, h = hw >> 6, w0 = hw & 63;
  float pnx = (n == 2) ? 1.f : 0.f;               // rows [0,0,1]
  float pny = (n == 1) ? 1.f : 0.f;               // cols [0,1,0]
  float ox = off[((size_t)(b * 6 + n) << 12) + hw];
  float oy = off[((size_t)(b * 6 + 3 + n) << 12) + hw];
  float px = (float)h + pnx + ox;
  float py = (float)w0 + pny + oy;
  float qx = floorf(px), qy = floorf(py);
  float qltx = clamp63(qx),       qlty = clamp63(qy);
  float qrbx = clamp63(qx + 1.f), qrby = clamp63(qy + 1.f);
  float pxc = clamp63(px), pyc = clamp63(py);
  float glt = (1.f + (qltx - pxc)) * (1.f + (qlty - pyc));
  float grb = (1.f - (qrbx - pxc)) * (1.f - (qrby - pyc));
  float glb = (1.f + (qltx - pxc)) * (1.f - (qrby - pyc));
  float grt = (1.f - (qrbx - pxc)) * (1.f + (qlty - pyc));
  int base = id * 4;
  idx[base + 0] = (int)qltx * 64 + (int)qlty;
  idx[base + 1] = (int)qrbx * 64 + (int)qrby;
  idx[base + 2] = (int)qltx * 64 + (int)qrby;
  idx[base + 3] = (int)qrbx * 64 + (int)qlty;
  wgt[base + 0] = glt; wgt[base + 1] = grb; wgt[base + 2] = glb; wgt[base + 3] = grt;
}

// ------------------------------------------------- gather -> x_off f16 [P][768]
__global__ __launch_bounds__(256) void xoff_kernel(
    const float* __restrict__ x, const int* __restrict__ idx,
    const float* __restrict__ wgt, _Float16* __restrict__ xo)
{
  int p = blockIdx.x, c = threadIdx.x;
  int b = p >> 12;
  const float* xb = x + (((size_t)b * CC + c) << 12);
#pragma unroll
  for (int n = 0; n < 3; ++n) {
    int base = (p * 3 + n) * 4;
    int i0 = idx[base], i1 = idx[base + 1], i2 = idx[base + 2], i3 = idx[base + 3];
    float w0 = wgt[base], w1 = wgt[base + 1], w2 = wgt[base + 2], w3 = wgt[base + 3];
    float v = w0 * xb[i0] + w1 * xb[i1] + w2 * xb[i2] + w3 * xb[i3];
    xo[(size_t)p * 768 + n * 256 + c] = (_Float16)v;
  }
}

// ---------------------------------------------------------- WMMA GEMM (fused)
// D[m,p] = sum_k A[m][k] * Bm[p][k]   (f16 in, f32 acc)
// B panel staged in LDS once per block via async global->LDS copies.
// mode 0: outf NCHW = acc
// mode 1: outh NHWC f16 = (acc + bias[m]) * umul[nchw]
// mode 2: outf NCHW = acc + bias[m] + resid[nchw]
__global__ __launch_bounds__(256) void gemm_kernel(
    const _Float16* __restrict__ A, const _Float16* __restrict__ Bm,
    int K, int mode, const float* __restrict__ bias, const float* __restrict__ umul,
    const float* __restrict__ resid, float* __restrict__ outf, _Float16* __restrict__ outh)
{
  extern __shared__ _Float16 Bs[];                // [16][K+8] halves, padded rows
  const int lane  = threadIdx.x & 31;
  const int wv    = threadIdx.x >> 5;             // 0..7 waves
  const int nlane = lane & 15;
  const int khalf = lane >> 4;                    // 0 or 1
  const int pbase = blockIdx.x * 16;
  const int p     = pbase + nlane;                // pixel (N dim)
  const int m0    = wv * 32;                      // two 16-row M tiles per wave
  const int ldsk  = K + 8;                        // padded row stride (halves)

  // ---- async copy of the 16 x K f16 B panel into LDS (16B chunks) ----
  {
    const int chunks_per_row = K >> 3;            // 8 halves per chunk
    const int nchunk = 16 * chunks_per_row;
    for (int cid = threadIdx.x; cid < nchunk; cid += 256) {
      int r   = cid / chunks_per_row;
      int col = (cid - r * chunks_per_row) << 3;
      unsigned lds_addr = (unsigned)(uintptr_t)(&Bs[r * ldsk + col]);
      const _Float16* gsrc = Bm + (size_t)(pbase + r) * K + col;
      asm volatile("global_load_async_to_lds_b128 %0, %1, off"
                   :: "v"(lds_addr), "v"((unsigned long long)(uintptr_t)gsrc)
                   : "memory");
    }
    asm volatile("s_wait_asynccnt 0x0" ::: "memory");
    __syncthreads();
  }

  v8f acc0 = {0.f, 0.f, 0.f, 0.f, 0.f, 0.f, 0.f, 0.f};
  v8f acc1 = acc0;
  const _Float16* bsrow = Bs + nlane * ldsk;
  const _Float16* arow0 = A + (size_t)(m0 + nlane) * K;
  const _Float16* arow1 = A + (size_t)(m0 + 16 + nlane) * K;

  for (int k0 = 0; k0 < K; k0 += 32) {
    const int ka = k0 + khalf * 8;                // 16-bit A/B lane layout
    v8h blo = *(const v8h*)(bsrow + ka);          // ds_load_b128
    v8h bhi = *(const v8h*)(bsrow + ka + 16);
    v16h bf = __builtin_shufflevector(blo, bhi, 0,1,2,3,4,5,6,7,8,9,10,11,12,13,14,15);
    v8h a0l = *(const v8h*)(arow0 + ka);
    v8h a0h = *(const v8h*)(arow0 + ka + 16);
    v16h af0 = __builtin_shufflevector(a0l, a0h, 0,1,2,3,4,5,6,7,8,9,10,11,12,13,14,15);
    v8h a1l = *(const v8h*)(arow1 + ka);
    v8h a1h = *(const v8h*)(arow1 + ka + 16);
    v16h af1 = __builtin_shufflevector(a1l, a1h, 0,1,2,3,4,5,6,7,8,9,10,11,12,13,14,15);
    __builtin_prefetch(arow0 + ka + 64, 0, 1);    // global_prefetch next A tile
    acc0 = __builtin_amdgcn_wmma_f32_16x16x32_f16(false, af0, false, bf, (short)0, acc0, false, false);
    acc1 = __builtin_amdgcn_wmma_f32_16x16x32_f16(false, af1, false, bf, (short)0, acc1, false, false);
  }

  const int b = p >> 12, hw = p & 4095;
#pragma unroll
  for (int r = 0; r < 8; ++r) {
#pragma unroll
    for (int t = 0; t < 2; ++t) {
      int m = m0 + t * 16 + khalf * 8 + r;        // C/D layout: VGPR r -> M = r + 8*khalf
      float v = (t == 0) ? acc0[r] : acc1[r];
      size_t nchw = (((size_t)b * CC + m) << 12) + hw;
      if (mode == 0) {
        outf[nchw] = v;
      } else if (mode == 1) {
        outh[(size_t)p * CC + m] = (_Float16)((v + bias[m]) * umul[nchw]);
      } else {
        outf[nchw] = v + bias[m] + resid[nchw];
      }
    }
  }
}

// -------------------------------------------------------- BN stats per channel
__global__ __launch_bounds__(256) void bnstats_kernel(
    const float* __restrict__ y, const float* __restrict__ gamma,
    const float* __restrict__ beta, float* __restrict__ stats)
{
  __shared__ float sh[512];
  int c = blockIdx.x, tid = threadIdx.x;
  float s = 0.f, s2 = 0.f;
  for (int b = 0; b < BB; ++b) {
    const float* pl = y + (((size_t)b * CC + c) << 12);
    for (int i = tid; i < HHW; i += 256) { float v = pl[i]; s += v; s2 += v * v; }
  }
  sh[tid] = s; sh[256 + tid] = s2;
  __syncthreads();
  for (int st = 128; st > 0; st >>= 1) {
    if (tid < st) { sh[tid] += sh[tid + st]; sh[256 + tid] += sh[256 + tid + st]; }
    __syncthreads();
  }
  if (tid == 0) {
    float mean = sh[0] * (1.f / 32768.f);
    float var  = sh[256] * (1.f / 32768.f) - mean * mean;
    float sc = gamma[c] * rsqrtf(var + 1e-5f);
    stats[c] = sc;
    stats[256 + c] = beta[c] - mean * sc;
  }
}

// ----------------------------------------------------------------- BN + SiLU
__global__ __launch_bounds__(256) void bnsilu_kernel(
    const float* __restrict__ y, const float* __restrict__ stats, float* __restrict__ u)
{
  int tid = blockIdx.x * 256 + threadIdx.x;
  int c = (tid >> 12) & 255;
  float v = y[tid] * stats[c] + stats[256 + c];
  u[tid] = v / (1.f + __expf(-v));
}

// --------------------------------------------------------- depthwise 5x5 pad2
__global__ __launch_bounds__(256) void dw5_kernel(
    const float* __restrict__ u, const float* __restrict__ w5,
    const float* __restrict__ b5, float* __restrict__ out)
{
  int tid = blockIdx.x * 256 + threadIdx.x;
  int w0 = tid & 63, h = (tid >> 6) & 63, c = (tid >> 12) & 255;
  const float* pl = u + ((size_t)(tid >> 12) << 12);
  const float* wp = w5 + c * 25;
  float acc = b5[c];
#pragma unroll
  for (int ky = 0; ky < 5; ++ky) {
    int yy = h + ky - 2;
    if (yy < 0 || yy > 63) continue;
#pragma unroll
    for (int kx = 0; kx < 5; ++kx) {
      int xx = w0 + kx - 2;
      if (xx < 0 || xx > 63) continue;
      acc = fmaf(pl[yy * 64 + xx], wp[ky * 5 + kx], acc);
    }
  }
  out[tid] = acc;
}

// --------------------------------------- depthwise 7x7 dil3 pad9 -> NHWC f16
__global__ __launch_bounds__(256) void dw7_kernel(
    const float* __restrict__ a1, const float* __restrict__ w7,
    const float* __restrict__ b7, _Float16* __restrict__ outh)
{
  int tid = blockIdx.x * 256 + threadIdx.x;
  int w0 = tid & 63, h = (tid >> 6) & 63, c = (tid >> 12) & 255, b = tid >> 20;
  const float* pl = a1 + ((size_t)(tid >> 12) << 12);
  const float* wp = w7 + c * 49;
  float acc = b7[c];
#pragma unroll
  for (int ky = 0; ky < 7; ++ky) {
    int yy = h + ky * 3 - 9;
    if (yy < 0 || yy > 63) continue;
#pragma unroll
    for (int kx = 0; kx < 7; ++kx) {
      int xx = w0 + kx * 3 - 9;
      if (xx < 0 || xx > 63) continue;
      acc = fmaf(pl[yy * 64 + xx], wp[ky * 7 + kx], acc);
    }
  }
  int p = (b << 12) + (tid & 4095);
  outh[(size_t)p * CC + c] = (_Float16)acc;
}

// ---------------------------------------------------------------------- host
extern "C" void kernel_launch(void* const* d_in, const int* in_sizes, int n_in,
                              void* d_out, int out_size, void* d_ws, size_t ws_size,
                              hipStream_t stream)
{
  const float* x    = (const float*)d_in[0];
  const float* p_w  = (const float*)d_in[1];
  const float* p_b  = (const float*)d_in[2];
  const float* ak_w = (const float*)d_in[3];
  const float* akg  = (const float*)d_in[4];
  const float* akb  = (const float*)d_in[5];
  const float* w5   = (const float*)d_in[6];
  const float* b5   = (const float*)d_in[7];
  const float* w7   = (const float*)d_in[8];
  const float* b7   = (const float*)d_in[9];
  const float* w1   = (const float*)d_in[10];
  const float* b1   = (const float*)d_in[11];
  const float* cw   = (const float*)d_in[12];
  const float* cb   = (const float*)d_in[13];
  float* out = (float*)d_out;
  char* ws = (char*)d_ws;

  const size_t o_off = 0;
  const size_t o_idx = o_off + (size_t)6  * PP * 4;     // 0.75 MB offsets
  const size_t o_wgt = o_idx + (size_t)12 * PP * 4;     // 1.5 MB idx
  const size_t o_st  = o_wgt + (size_t)12 * PP * 4;     // 1.5 MB wgt
  const size_t o_Aak = o_st  + 2048;
  const size_t o_Alk = o_Aak + (size_t)256 * 768 * 2;
  const size_t o_Acv = o_Alk + (size_t)256 * 256 * 2;
  const size_t o_u   = o_Acv + (size_t)256 * 256 * 2;
  const size_t o_y1  = o_u   + (size_t)CC * PP * 4;     // 32 MB u
  const size_t o_x0  = o_y1  + (size_t)CC * PP * 4;     // 32 MB y1 (reused as dw5 out)
  // o_x0: 48 MB x_off f16; reused afterwards as attn (16 MB) + y2 (16 MB)

  float*    off   = (float*)(ws + o_off);
  int*      idx   = (int*)(ws + o_idx);
  float*    wgt   = (float*)(ws + o_wgt);
  float*    stats = (float*)(ws + o_st);
  _Float16* Aak   = (_Float16*)(ws + o_Aak);
  _Float16* Alk   = (_Float16*)(ws + o_Alk);
  _Float16* Acv   = (_Float16*)(ws + o_Acv);
  float*    ubuf  = (float*)(ws + o_u);
  float*    y1    = (float*)(ws + o_y1);
  float*    a1    = y1;                                  // reuse after BN consumed y1
  _Float16* xo    = (_Float16*)(ws + o_x0);
  _Float16* attn  = (_Float16*)(ws + o_x0);              // reuse x_off region
  _Float16* y2h   = (_Float16*)(ws + o_x0 + (size_t)16 * 1024 * 1024);

  repack_kernel<<<768, 256, 0, stream>>>(ak_w, w1, cw, Aak, Alk, Acv);
  pconv_kernel<<<PP / 256, 256, 0, stream>>>(x, p_w, p_b, off);
  sprep_kernel<<<(3 * PP) / 256, 256, 0, stream>>>(off, idx, wgt);
  xoff_kernel<<<PP, 256, 0, stream>>>(x, idx, wgt, xo);
  gemm_kernel<<<PP / 16, 256, (size_t)16 * (KAK + 8) * 2, stream>>>(
      Aak, xo, KAK, 0, nullptr, nullptr, nullptr, y1, nullptr);
  bnstats_kernel<<<CC, 256, 0, stream>>>(y1, akg, akb, stats);
  bnsilu_kernel<<<(CC * PP) / 256, 256, 0, stream>>>(y1, stats, ubuf);
  dw5_kernel<<<(CC * PP) / 256, 256, 0, stream>>>(ubuf, w5, b5, a1);
  dw7_kernel<<<(CC * PP) / 256, 256, 0, stream>>>(a1, w7, b7, attn);
  gemm_kernel<<<PP / 16, 256, (size_t)16 * (256 + 8) * 2, stream>>>(
      Alk, attn, 256, 1, b1, ubuf, nullptr, nullptr, y2h);
  gemm_kernel<<<PP / 16, 256, (size_t)16 * (256 + 8) * 2, stream>>>(
      Acv, y2h, 256, 2, cb, nullptr, x, out, nullptr);
}
